// self_attention_11115375362081
// MI455X (gfx1250) — matrix operand order repaired and checked by reading, not compiled, not run on hardware
//
#include <hip/hip_runtime.h>

typedef __bf16 bf16x16 __attribute__((ext_vector_type(16)));
typedef float f32x8 __attribute__((ext_vector_type(8)));
typedef unsigned int u32x4 __attribute__((ext_vector_type(4)));
typedef unsigned int v4u __attribute__((ext_vector_type(4)));
typedef int v4i __attribute__((ext_vector_type(4)));
typedef int v8i __attribute__((ext_vector_type(8)));

union FragBF {
  bf16x16 v;
  u32x4 q[2];
};

__device__ __forceinline__ unsigned short f32_bf16(float f) {
  unsigned u = __builtin_bit_cast(unsigned, f);
  u += 0x7FFFu + ((u >> 16) & 1u);   // round-to-nearest-even
  return (unsigned short)(u >> 16);
}

__device__ __forceinline__ f32x8 wmma_bf16(const FragBF& a, const FragBF& b, f32x8 c) {
  return __builtin_amdgcn_wmma_f32_16x16x32_bf16(false, a.v, false, b.v, (short)0, c,
                                                 false, false);
}

// gfx1250 async global->LDS copy (ASYNCcnt tracked), 16B per lane.
__device__ __forceinline__ void async_copy_b128(unsigned lds_byte_off,
                                                const unsigned short* g) {
  asm volatile("global_load_async_to_lds_b128 %0, %1, off"
               :
               : "v"(lds_byte_off), "v"(g)
               : "memory");
}
__device__ __forceinline__ void wait_async0() {
  asm volatile("s_wait_asynccnt 0" ::: "memory");
}

// gfx1250 Tensor Data Mover: DMA a 2D tile (tile_d0 x tile_d1 elements of 2 bytes)
// from a row-major tensor (row stride = td0_stride elems) into LDS.
// Descriptor bitfields per CDNA5 ISA 8.3/8.4. Issue from ONE thread.
__device__ __forceinline__ void tdm_load_2d_b16(unsigned lds_off, const void* gaddr,
                                                unsigned tensor_d0, unsigned tensor_d1,
                                                unsigned tile_d0, unsigned tile_d1,
                                                unsigned td0_stride) {
#if __has_builtin(__builtin_amdgcn_tensor_load_to_lds)
  unsigned long long ga = (unsigned long long)gaddr;
  v4u g0;
  g0[0] = 1u;                                   // count=1, user mode
  g0[1] = lds_off;                              // lds_addr
  g0[2] = (unsigned)(ga & 0xFFFFFFFFu);         // global_addr[31:0]
  g0[3] = (unsigned)((ga >> 32) & 0x01FFFFFFu)  // global_addr[56:32]
          | (2u << 30);                         // type=2 ("image")
  v8i g1;
  g1[0] = (int)(1u << 16);                      // data_size=1 -> 2 bytes
  g1[1] = (int)(tensor_d0 << 16);               // tensor_dim0[15:0] @bits[63:48]
  g1[2] = (int)((tensor_d0 >> 16) | (tensor_d1 << 16));  // dim0 hi | dim1 lo
  g1[3] = (int)((tensor_d1 >> 16) | (tile_d0 << 16));    // dim1 hi | tile_dim0
  g1[4] = (int)(tile_d1 & 0xFFFFu);             // tile_dim1 (tile_dim2=0)
  g1[5] = (int)td0_stride;                      // tensor_dim0_stride[31:0]
  g1[6] = 0;                                    // stride hi | dim1_stride lo
  g1[7] = 0;
  v4i z = {0, 0, 0, 0};
#if defined(__clang_major__) && __clang_major__ >= 23
  v8i z8 = {0, 0, 0, 0, 0, 0, 0, 0};
  __builtin_amdgcn_tensor_load_to_lds(g0, g1, z, z, z8, 0);
#else
  __builtin_amdgcn_tensor_load_to_lds(g0, g1, z, z, 0);
#endif
  __builtin_amdgcn_s_wait_tensorcnt(0);
#endif
}

#define CIN 512
#define HWD 1024
#define QSCALE 0.17677669529663687f  // 32^-0.5

// ---------------- Kernel 1: QKV projection GEMM (M=768, N=1024, K=512, 8 batches)
// Writes Q,K as (b, head, hw, 32) bf16 (Q pre-scaled), V as (b, head, 32, hw) bf16.
__global__ __launch_bounds__(256) void qkv_gemm(
    const float* __restrict__ x, const float* __restrict__ w,
    const float* __restrict__ bias, unsigned short* __restrict__ Qw,
    unsigned short* __restrict__ Kw, unsigned short* __restrict__ Vw) {
  __shared__ __align__(16) unsigned short As[64 * 32];
  __shared__ __align__(16) unsigned short Bs[128 * 32];

  const int t = threadIdx.x;
  const int lane = t & 31, wave = t >> 5;
  const int laneRow = lane & 15, half = lane >> 4;
  const int wm = wave & 1, wn = wave >> 1;
  const int bb = blockIdx.z;
  const int m0 = blockIdx.y * 64;
  const int n0 = blockIdx.x * 128;
  const float* xb = x + (size_t)bb * CIN * HWD;

  f32x8 acc[2][2] = {};

  for (int k0 = 0; k0 < CIN; k0 += 32) {
    // stage A: w[m0..+64)[k0..+32) fp32 -> bf16, row-major
    {
      int row = t >> 2;
      int col = (t & 3) * 8;
      const float* src = w + (size_t)(m0 + row) * CIN + k0 + col;
      if (k0 + 32 < CIN) __builtin_prefetch(src + 32, 0, 3);
#pragma unroll
      for (int j = 0; j < 8; ++j) As[row * 32 + col + j] = f32_bf16(src[j]);
    }
    // stage B transposed: x[k0..+32)[n0..+128) -> Bs[hw_local][c_local]
#pragma unroll
    for (int i = 0; i < 16; ++i) {
      int idx = i * 256 + t;
      int cl = idx >> 7;
      int hwl = idx & 127;
      Bs[hwl * 32 + cl] = f32_bf16(xb[(size_t)(k0 + cl) * HWD + n0 + hwl]);
    }
    __syncthreads();

    FragBF a[2], b[2];
#pragma unroll
    for (int sm = 0; sm < 2; ++sm) {
      int row = wm * 32 + sm * 16 + laneRow;
      a[sm].q[0] = *(const u32x4*)&As[row * 32 + half * 8];
      a[sm].q[1] = *(const u32x4*)&As[row * 32 + 16 + half * 8];
    }
#pragma unroll
    for (int sn = 0; sn < 2; ++sn) {
      int row = wn * 32 + sn * 16 + laneRow;
      b[sn].q[0] = *(const u32x4*)&Bs[row * 32 + half * 8];
      b[sn].q[1] = *(const u32x4*)&Bs[row * 32 + 16 + half * 8];
    }
#pragma unroll
    for (int sm = 0; sm < 2; ++sm)
#pragma unroll
      for (int sn = 0; sn < 2; ++sn) acc[sm][sn] = wmma_bf16(a[sm], b[sn], acc[sm][sn]);
    __syncthreads();
  }

#pragma unroll
  for (int sm = 0; sm < 2; ++sm)
#pragma unroll
    for (int sn = 0; sn < 2; ++sn)
#pragma unroll
      for (int r = 0; r < 8; ++r) {
        int o = m0 + wm * 32 + sm * 16 + r + half * 8;
        int hw = n0 + wn * 32 + sn * 16 + laneRow;
        float v = acc[sm][sn][r] + bias[o];
        if (o < 256) {
          int head = o >> 5, d = o & 31;
          Qw[(((size_t)bb * 8 + head) * HWD + hw) * 32 + d] = f32_bf16(v * QSCALE);
        } else if (o < 512) {
          int o2 = o - 256, head = o2 >> 5, d = o2 & 31;
          Kw[(((size_t)bb * 8 + head) * HWD + hw) * 32 + d] = f32_bf16(v);
        } else {
          int o2 = o - 512, head = o2 >> 5, d = o2 & 31;
          Vw[(((size_t)bb * 8 + head) * 32 + d) * HWD + hw] = f32_bf16(v);
        }
      }
}

// ---------------- Kernel 2: fused attention. One wave per (batch, 16-row q-tile).
// logits for all 8 heads -> softmax across heads (per lane) -> AV accumulation.
// Stores attn in the reference's reshape order as (b, hw', c=256) bf16.
__global__ __launch_bounds__(32) void attn_fused(
    const unsigned short* __restrict__ Qw, const unsigned short* __restrict__ Kw,
    const unsigned short* __restrict__ Vw, unsigned short* __restrict__ AT) {
  __shared__ __align__(16) unsigned short Wl[16 * 32];

  const int lane = threadIdx.x & 31;
  const int laneRow = lane & 15, half = lane >> 4;
  const int gw = blockIdx.x;  // 0..511
  const int bb = gw >> 6;
  const int q0 = (gw & 63) * 16;

  FragBF qf[8];
#pragma unroll
  for (int h = 0; h < 8; ++h) {
    const unsigned short* qp = Qw + (((size_t)bb * 8 + h) * HWD + q0 + laneRow) * 32;
    qf[h].q[0] = *(const u32x4*)(qp + half * 8);
    qf[h].q[1] = *(const u32x4*)(qp + 16 + half * 8);
  }

  f32x8 acc[8][2] = {};

  for (int k0 = 0; k0 < HWD; k0 += 32) {
    f32x8 lg[8][2];
#pragma unroll
    for (int h = 0; h < 8; ++h)
#pragma unroll
      for (int s = 0; s < 2; ++s) {
        FragBF kb;
        const unsigned short* kp =
            Kw + (((size_t)bb * 8 + h) * HWD + k0 + s * 16 + laneRow) * 32 + half * 8;
        kb.q[0] = *(const u32x4*)(kp);
        kb.q[1] = *(const u32x4*)(kp + 16);
        f32x8 z = {};
        lg[h][s] = wmma_bf16(qf[h], kb, z);
      }
    // softmax over the HEAD axis (reference: softmax(logits, axis=1))
#pragma unroll
    for (int s = 0; s < 2; ++s)
#pragma unroll
      for (int r = 0; r < 8; ++r) {
        float m = lg[0][s][r];
#pragma unroll
        for (int h = 1; h < 8; ++h) m = fmaxf(m, lg[h][s][r]);
        float sum = 0.f;
#pragma unroll
        for (int h = 0; h < 8; ++h) {
          float e = __expf(lg[h][s][r] - m);
          lg[h][s][r] = e;
          sum += e;
        }
        float inv = __builtin_amdgcn_rcpf(sum);
#pragma unroll
        for (int h = 0; h < 8; ++h) lg[h][s][r] *= inv;
      }
    // per head: D-layout weights -> LDS (16x32 row-major) -> A-frag -> AV WMMA
#pragma unroll
    for (int h = 0; h < 8; ++h) {
#pragma unroll
      for (int s = 0; s < 2; ++s)
#pragma unroll
        for (int r = 0; r < 8; ++r)
          Wl[(r + half * 8) * 32 + s * 16 + laneRow] = f32_bf16(lg[h][s][r]);
      asm volatile("s_wait_dscnt 0" ::: "memory");
      FragBF wf;
      wf.q[0] = *(const u32x4*)&Wl[laneRow * 32 + half * 8];
      wf.q[1] = *(const u32x4*)&Wl[laneRow * 32 + 16 + half * 8];
#pragma unroll
      for (int dt = 0; dt < 2; ++dt) {
        FragBF vb;
        const unsigned short* vp =
            Vw + (((size_t)bb * 8 + h) * 32 + dt * 16 + laneRow) * HWD + k0 + half * 8;
        vb.q[0] = *(const u32x4*)(vp);
        vb.q[1] = *(const u32x4*)(vp + 16);
        acc[h][dt] = wmma_bf16(wf, vb, acc[h][dt]);
      }
    }
  }

  // reference reshape: flat (q*32+d) -> channel d'=q>>5, pos hw'=(q&31)*32+d
#pragma unroll
  for (int h = 0; h < 8; ++h)
#pragma unroll
    for (int dt = 0; dt < 2; ++dt)
#pragma unroll
      for (int r = 0; r < 8; ++r) {
        int q = q0 + r + half * 8;
        int d = dt * 16 + laneRow;
        int c = h * 32 + (q >> 5);
        int hw2 = ((q & 31) << 5) | d;
        AT[((size_t)bb * HWD + hw2) * 256 + c] = f32_bf16(acc[h][dt][r]);
      }
}

// ---------------- Kernel 3: output projection GEMM (M=512, N=1024, K=256)
// B tile is already bf16 in the right layout. Alternate TDM tensor DMA (even
// K-steps) with per-lane async global->LDS (odd K-steps).
__global__ __launch_bounds__(256) void out_gemm(
    const unsigned short* __restrict__ AT, const float* __restrict__ w,
    const float* __restrict__ bias, float* __restrict__ out) {
  __shared__ __align__(16) unsigned short As[64 * 32];
  __shared__ __align__(16) unsigned short Bs[128 * 32];

  const int t = threadIdx.x;
  const int lane = t & 31, wave = t >> 5;
  const int laneRow = lane & 15, half = lane >> 4;
  const int wm = wave & 1, wn = wave >> 1;
  const int bb = blockIdx.z;
  const int m0 = blockIdx.y * 64;
  const int n0 = blockIdx.x * 128;

  f32x8 acc[2][2] = {};

  for (int k0 = 0; k0 < 256; k0 += 32) {
    const unsigned short* btile = AT + ((size_t)bb * HWD + n0) * 256 + k0;
#if __has_builtin(__builtin_amdgcn_tensor_load_to_lds)
    const bool use_tdm = (k0 & 32) == 0;
#else
    const bool use_tdm = false;
#endif
    if (use_tdm) {
      if (t == 0)  // one TDM descriptor moves the whole 128x32 bf16 tile
        tdm_load_2d_b16((unsigned)(uintptr_t)&Bs[0], btile,
                        /*tensor_d0=*/256, /*tensor_d1=*/HWD,
                        /*tile_d0=*/32, /*tile_d1=*/128, /*td0_stride=*/256);
    } else {
#pragma unroll
      for (int i = 0; i < 2; ++i) {
        int uidx = i * 256 + t;  // 512 x 16B chunks
        int row = uidx >> 2;
        int c8 = (uidx & 3) * 8;
        async_copy_b128((unsigned)(uintptr_t)&Bs[row * 32 + c8],
                        btile + (size_t)row * 256 + c8);
      }
    }
    // stage A with fp32->bf16 conversion (overlaps with the DMA)
    {
      int row = t >> 2;
      int col = (t & 3) * 8;
      const float* src = w + (size_t)(m0 + row) * 256 + k0 + col;
#pragma unroll
      for (int j = 0; j < 8; ++j) As[row * 32 + col + j] = f32_bf16(src[j]);
    }
    if (!use_tdm) wait_async0();
    __syncthreads();

    FragBF a[2], b[2];
#pragma unroll
    for (int sm = 0; sm < 2; ++sm) {
      int row = wm * 32 + sm * 16 + laneRow;
      a[sm].q[0] = *(const u32x4*)&As[row * 32 + half * 8];
      a[sm].q[1] = *(const u32x4*)&As[row * 32 + 16 + half * 8];
    }
#pragma unroll
    for (int sn = 0; sn < 2; ++sn) {
      int row = wn * 32 + sn * 16 + laneRow;
      b[sn].q[0] = *(const u32x4*)&Bs[row * 32 + half * 8];
      b[sn].q[1] = *(const u32x4*)&Bs[row * 32 + 16 + half * 8];
    }
#pragma unroll
    for (int sm = 0; sm < 2; ++sm)
#pragma unroll
      for (int sn = 0; sn < 2; ++sn) acc[sm][sn] = wmma_bf16(a[sm], b[sn], acc[sm][sn]);
    __syncthreads();
  }

#pragma unroll
  for (int sm = 0; sm < 2; ++sm)
#pragma unroll
    for (int sn = 0; sn < 2; ++sn)
#pragma unroll
      for (int r = 0; r < 8; ++r) {
        int o = m0 + wm * 32 + sm * 16 + r + half * 8;
        int hw = n0 + wn * 32 + sn * 16 + laneRow;
        out[((size_t)bb * 512 + o) * HWD + hw] = acc[sm][sn][r] + bias[o];
      }
}

extern "C" void kernel_launch(void* const* d_in, const int* in_sizes, int n_in,
                              void* d_out, int out_size, void* d_ws, size_t ws_size,
                              hipStream_t stream) {
  const float* x = (const float*)d_in[0];
  const float* w_qkv = (const float*)d_in[1];
  const float* b_qkv = (const float*)d_in[2];
  const float* w_attn = (const float*)d_in[3];
  const float* b_attn = (const float*)d_in[4];
  float* out = (float*)d_out;

  // workspace: Q,K,V = (8,8,1024,32)/(...)/(8,8,32,1024) bf16, AT = (8,1024,256) bf16
  const size_t SEG = (size_t)8 * 8 * 1024 * 32;  // 2 M elems = 4 MB each
  unsigned short* Qw = (unsigned short*)d_ws;
  unsigned short* Kw = Qw + SEG;
  unsigned short* Vw = Kw + SEG;
  unsigned short* AT = Vw + SEG;

  qkv_gemm<<<dim3(8, 12, 8), 256, 0, stream>>>(x, w_qkv, b_qkv, Qw, Kw, Vw);
  attn_fused<<<dim3(512), 32, 0, stream>>>(Qw, Kw, Vw, AT);
  out_gemm<<<dim3(8, 8, 8), 256, 0, stream>>>(AT, w_attn, b_attn, out);
}
